// MHA_73023033966818
// MI455X (gfx1250) — matrix operand order repaired and checked by reading, not compiled, not run on hardware
//
#include <hip/hip_runtime.h>
#include <hip/hip_bf16.h>

// ---------------------------------------------------------------------------
// MI455X (gfx1250) fused MHA: QKV proj -> RoPE+RMSNorm -> flash attn -> out proj
// bf16 WMMA math everywhere; tiles staged with GLOBAL_LOAD_ASYNC_TO_LDS_B128
// (ASYNCcnt-pipelined, double-buffered in the GEMM).
// ---------------------------------------------------------------------------

typedef __attribute__((ext_vector_type(16))) __bf16  bf16x16;
typedef __attribute__((ext_vector_type(8)))  float   f32x8;
typedef __attribute__((ext_vector_type(4)))  unsigned int u32x4;

union Frag {
    bf16x16 v;
    u32x4   q[2];
    unsigned int u[8];
};

__device__ __forceinline__ unsigned short f2bf(float f) {
    unsigned int u = __float_as_uint(f);
    u += 0x7FFFu + ((u >> 16) & 1u);   // round-to-nearest-even
    return (unsigned short)(u >> 16);
}

__device__ __forceinline__ f32x8 wmma_bf16(const Frag& a, const Frag& b, f32x8 c) {
    return __builtin_amdgcn_wmma_f32_16x16x32_bf16(
        false, a.v, false, b.v, (short)0, c, false, false);
}

__device__ __forceinline__ void load_frag_split(Frag& f, const unsigned short* p0,
                                                const unsigned short* p1) {
    f.q[0] = *(const u32x4*)p0;
    f.q[1] = *(const u32x4*)p1;
}
__device__ __forceinline__ void load_frag_contig(Frag& f, const unsigned short* p) {
    f.q[0] = *(const u32x4*)p;
    f.q[1] = *(const u32x4*)(p + 8);
}

// generic pointer to an LDS object -> 32-bit LDS offset (flat addr low bits)
__device__ __forceinline__ unsigned int lds_off(const void* p) {
    return (unsigned int)(unsigned long long)p;
}

// async 16B global -> LDS copy, per lane (tracked by ASYNCcnt)
__device__ __forceinline__ void async_b128(unsigned int lds, const void* gaddr) {
    asm volatile("global_load_async_to_lds_b128 %0, %1, off"
                 :: "v"(lds), "v"(gaddr) : "memory");
}
__device__ __forceinline__ void wait_async_0() {
    asm volatile("s_wait_asynccnt 0x0" ::: "memory");
}
__device__ __forceinline__ void wait_async_6() {
    asm volatile("s_wait_asynccnt 0x6" ::: "memory");
}

// ---------------------------------------------------------------------------
// One-time fp32 -> bf16 convert (x) and convert+transpose (weights, V).
// ---------------------------------------------------------------------------
__global__ void __launch_bounds__(256)
convert_f32_to_bf16(const float* __restrict__ in, unsigned short* __restrict__ out, int n) {
    for (int i = blockIdx.x * 256 + threadIdx.x; i < n; i += gridDim.x * 256)
        out[i] = f2bf(in[i]);
}

// in: fp32 [R][C]  ->  out: bf16 [C][R]   (R, C multiples of 64)
__global__ void __launch_bounds__(256)
transpose_f32_to_bf16(const float* __restrict__ in, unsigned short* __restrict__ out,
                      int R, int C) {
    __shared__ unsigned short t[64][65];
    const int r0 = blockIdx.y * 64, c0 = blockIdx.x * 64;
    for (int i = threadIdx.x; i < 64 * 64; i += 256) {
        int r = i >> 6, c = i & 63;
        t[c][r] = f2bf(in[(size_t)(r0 + r) * C + c0 + c]);
    }
    __syncthreads();
    for (int i = threadIdx.x; i < 64 * 64; i += 256) {
        int c = i >> 6, r = i & 63;
        out[(size_t)(c0 + c) * R + r0 + r] = t[c][r];
    }
}

// ---------------------------------------------------------------------------
// GEMM: C[M,N](fp32) = A[M,K](bf16 row-major) * Bt[N,K](bf16, pre-transposed).
// Block 128 thr (4 waves), block tile 128x64, wave tile 32x64, K step 32.
// Tiles staged with async-to-LDS, double buffered (6 async instr / wave / tile).
// ---------------------------------------------------------------------------
#define GTM 128
#define GTN 64
#define GTK 32

__global__ void __launch_bounds__(128)
wmma_gemm_bf16(const unsigned short* __restrict__ A,
               const unsigned short* __restrict__ Bt,
               float* __restrict__ C, int M, int N, int K) {
    __shared__ unsigned short sA[2][GTM * GTK];   // [m][k]
    __shared__ unsigned short sB[2][GTN * GTK];   // [n][k]

    const int tid  = threadIdx.x;
    const int lane = tid & 31;
    const int wave = tid >> 5;
    const int half = lane >> 4;
    const int l16  = lane & 15;
    const int m0 = blockIdx.y * GTM;
    const int n0 = blockIdx.x * GTN;

    f32x8 acc[8];                                  // [strip 0/1][4 col tiles]
    #pragma unroll
    for (int t = 0; t < 8; ++t)
        for (int j = 0; j < 8; ++j) acc[t][j] = 0.f;

    const int nkt = K / GTK;

    auto issue = [&](int t, int b) {
        const int k0 = t * GTK;
        #pragma unroll
        for (int j = 0; j < 4; ++j) {              // A: 128x32 = 512 chunks
            int ci = tid + j * 128;
            int r = ci >> 2, c = ci & 3;
            async_b128(lds_off(&sA[b][r * GTK + c * 8]),
                       A + (size_t)(m0 + r) * K + k0 + c * 8);
        }
        #pragma unroll
        for (int j = 0; j < 2; ++j) {              // B: 64x32 = 256 chunks
            int ci = tid + j * 128;
            int r = ci >> 2, c = ci & 3;
            async_b128(lds_off(&sB[b][r * GTK + c * 8]),
                       Bt + (size_t)(n0 + r) * K + k0 + c * 8);
        }
    };

    issue(0, 0);
    for (int t = 0; t < nkt; ++t) {
        const int b = t & 1;
        if (t + 1 < nkt) { issue(t + 1, b ^ 1); wait_async_6(); }
        else             { wait_async_0(); }
        __syncthreads();

        Frag a0, a1;
        const int r0 = wave * 32 + l16;
        const int r1 = r0 + 16;
        load_frag_split(a0, &sA[b][r0 * GTK + half * 8], &sA[b][r0 * GTK + 16 + half * 8]);
        load_frag_split(a1, &sA[b][r1 * GTK + half * 8], &sA[b][r1 * GTK + 16 + half * 8]);

        #pragma unroll
        for (int nt = 0; nt < 4; ++nt) {
            Frag bb;
            load_frag_split(bb, &sB[b][(nt * 16 + l16) * GTK + half * 8], &sB[b][(nt * 16 + l16) * GTK + 16 + half * 8]);
            acc[nt]     = wmma_bf16(a0, bb, acc[nt]);
            acc[4 + nt] = wmma_bf16(a1, bb, acc[4 + nt]);
        }
        __syncthreads();
    }

    #pragma unroll
    for (int s = 0; s < 2; ++s)
        #pragma unroll
        for (int nt = 0; nt < 4; ++nt)
            #pragma unroll
            for (int i = 0; i < 8; ++i) {
                int m = m0 + wave * 32 + s * 16 + half * 8 + i;
                int n = n0 + nt * 16 + l16;
                C[(size_t)m * N + n] = acc[s * 4 + nt][i];
            }
}

// ---------------------------------------------------------------------------
// RoPE + RMSNorm, one wave per (token, head). fp32 in -> bf16 out.
// ---------------------------------------------------------------------------
__global__ void __launch_bounds__(128)
rope_rms_kernel(const float* __restrict__ src, const float* __restrict__ cs,
                const float* __restrict__ sn, unsigned short* __restrict__ dst,
                int nheads) {
    const int wave = threadIdx.x >> 5;
    const int lane = threadIdx.x & 31;
    const int item = blockIdx.x * 4 + wave;
    const int t = item / nheads;
    const int h = item % nheads;

    const float* row = src + (size_t)t * nheads * 128 + h * 128;
    float y[4];
    float ss = 0.f;
    #pragma unroll
    for (int u = 0; u < 2; ++u) {
        int p = lane + 32 * u;
        float c = cs[t * 64 + p], s = sn[t * 64 + p];
        float x1 = row[p], x2 = row[p + 64];
        float y1 = x1 * c + x2 * s;
        float y2 = x2 * c - x1 * s;
        y[2 * u] = y1; y[2 * u + 1] = y2;
        ss += y1 * y1 + y2 * y2;
    }
    #pragma unroll
    for (int o = 16; o > 0; o >>= 1) ss += __shfl_xor(ss, o, 32);
    const float r = rsqrtf(ss * (1.f / 128.f) + 1e-6f);

    unsigned short* orow = dst + (size_t)t * nheads * 128 + h * 128;
    #pragma unroll
    for (int u = 0; u < 2; ++u) {
        int p = lane + 32 * u;
        orow[p]      = f2bf(y[2 * u] * r);
        orow[p + 64] = f2bf(y[2 * u + 1] * r);
    }
}

// ---------------------------------------------------------------------------
// Flash attention (causal, GQA 4:1). Block: 4 waves = 64 query rows, one head.
// K tile [32 keys][128d] and V tile [128d][32 keys] staged with async-to-LDS.
// Qn/Kn bf16 roped+normed; Vt bf16 pre-transposed [kvh*128+d][token].
// Output: bf16 Y[token][2048].
// ---------------------------------------------------------------------------
#define QB 64
#define KB 32
#define HD 128

__global__ void __launch_bounds__(128)
flash_attn_kernel(const unsigned short* __restrict__ Qn,
                  const unsigned short* __restrict__ Kn,
                  const unsigned short* __restrict__ Vt,
                  unsigned short* __restrict__ Y) {
    __shared__ unsigned short sK[KB * HD];        // [key][d]
    __shared__ unsigned short sVt[HD * KB];       // [d][key]
    __shared__ unsigned short sP[4][16 * KB];     // per-wave P scratch

    const int tid  = threadIdx.x;
    const int lane = tid & 31;
    const int wave = tid >> 5;
    const int half = lane >> 4;
    const int l16  = lane & 15;
    const int h    = blockIdx.y;
    const int kvh  = h >> 2;
    const int qb   = blockIdx.x * QB;
    const int qrow = qb + wave * 16 + l16;

    Frag qf[4];
    const unsigned short* qp = Qn + (size_t)qrow * 2048 + h * HD;
    #pragma unroll
    for (int c = 0; c < 4; ++c)
        load_frag_split(qf[c], qp + c * 32 + half * 8, qp + c * 32 + 16 + half * 8);

    f32x8 o[8];
    #pragma unroll
    for (int g = 0; g < 8; ++g)
        for (int j = 0; j < 8; ++j) o[g][j] = 0.f;
    float mrow[8], lrow[8];
    #pragma unroll
    for (int i = 0; i < 8; ++i) { mrow[i] = -3.0e38f; lrow[i] = 0.f; }

    const float scale = 0.088388347648318447f;    // 1/sqrt(128)
    const int nkt = (qb + QB) / KB;

    for (int kt = 0; kt < nkt; ++kt) {
        const int kbase = kt * KB;
        // K tile: 32x128 bf16 = 512 x 16B chunks, 4 per thread
        #pragma unroll
        for (int j = 0; j < 4; ++j) {
            int ci = tid + j * 128;
            int key = ci >> 4, ch = ci & 15;
            async_b128(lds_off(&sK[key * HD + ch * 8]),
                       Kn + (size_t)(kbase + key) * 512 + kvh * HD + ch * 8);
        }
        // V tile (already transposed in global): 128x32 bf16, 4 chunks/thread
        #pragma unroll
        for (int j = 0; j < 4; ++j) {
            int ci = tid + j * 128;
            int d = ci >> 2, ch = ci & 3;
            async_b128(lds_off(&sVt[d * KB + ch * 8]),
                       Vt + (size_t)(kvh * HD + d) * 4096 + kbase + ch * 8);
        }
        wait_async_0();
        __syncthreads();

        // S = Q * K^T : two 16x16 n-tiles, K-dim 128 = 4 x 32
        f32x8 s[2];
        #pragma unroll
        for (int nt = 0; nt < 2; ++nt) {
            for (int j = 0; j < 8; ++j) s[nt][j] = 0.f;
            #pragma unroll
            for (int c = 0; c < 4; ++c) {
                Frag b;
                load_frag_split(b, &sK[(nt * 16 + l16) * HD + c * 32 + half * 8], &sK[(nt * 16 + l16) * HD + c * 32 + 16 + half * 8]);
                s[nt] = wmma_bf16(qf[c], b, s[nt]);
            }
        }

        // scale + causal mask + online softmax (row lives in a 16-lane half)
        #pragma unroll
        for (int i = 0; i < 8; ++i) {
            const int m = qb + wave * 16 + half * 8 + i;
            #pragma unroll
            for (int nt = 0; nt < 2; ++nt) {
                int kc = kbase + nt * 16 + l16;
                float v = s[nt][i] * scale;
                if (kc > m) v = -3.0e38f;
                s[nt][i] = v;
            }
            float mx = fmaxf(s[0][i], s[1][i]);
            #pragma unroll
            for (int d = 8; d > 0; d >>= 1) mx = fmaxf(mx, __shfl_xor(mx, d, 32));
            float mnew = fmaxf(mrow[i], mx);
            float alpha = __expf(mrow[i] - mnew);
            mrow[i] = mnew;
            float rs = 0.f;
            #pragma unroll
            for (int nt = 0; nt < 2; ++nt) {
                float p = __expf(s[nt][i] - mnew);
                s[nt][i] = p;
                rs += p;
            }
            #pragma unroll
            for (int d = 8; d > 0; d >>= 1) rs += __shfl_xor(rs, d, 32);
            lrow[i] = lrow[i] * alpha + rs;
            #pragma unroll
            for (int g = 0; g < 8; ++g) o[g][i] *= alpha;
        }

        // C-layout P -> LDS -> A-fragment (per-wave buffer)
        unsigned short* pw = sP[wave];
        #pragma unroll
        for (int i = 0; i < 8; ++i) {
            int m = half * 8 + i;
            pw[m * KB + l16]      = f2bf(s[0][i]);
            pw[m * KB + 16 + l16] = f2bf(s[1][i]);
        }
        Frag pf;
        load_frag_split(pf, pw + l16 * KB + half * 8, pw + l16 * KB + 16 + half * 8);

        // O += P * V : 8 output dim-tiles of 16
        #pragma unroll
        for (int g = 0; g < 8; ++g) {
            Frag b;
            load_frag_split(b, &sVt[(g * 16 + l16) * KB + half * 8], &sVt[(g * 16 + l16) * KB + 16 + half * 8]);
            o[g] = wmma_bf16(pf, b, o[g]);
        }
        __syncthreads();
    }

    // epilogue: O /= l, write bf16 Y[token][h*128 + d]
    #pragma unroll
    for (int g = 0; g < 8; ++g)
        #pragma unroll
        for (int i = 0; i < 8; ++i) {
            int m = qb + wave * 16 + half * 8 + i;
            int n = h * HD + g * 16 + l16;
            Y[(size_t)m * 2048 + n] = f2bf(o[g][i] / lrow[i]);
        }
}

// ---------------------------------------------------------------------------
// Launch sequence.
// ---------------------------------------------------------------------------
extern "C" void kernel_launch(void* const* d_in, const int* in_sizes, int n_in,
                              void* d_out, int out_size, void* d_ws, size_t ws_size,
                              hipStream_t stream) {
    (void)in_sizes; (void)n_in; (void)out_size; (void)ws_size;
    const int T = 4096, C = 2048, NKV = 512;

    const float* x  = (const float*)d_in[0];
    const float* cs = (const float*)d_in[1];
    const float* sn = (const float*)d_in[2];
    const float* wq = (const float*)d_in[3];
    const float* wk = (const float*)d_in[4];
    const float* wv = (const float*)d_in[5];
    const float* wo = (const float*)d_in[6];
    float* out = (float*)d_out;

    unsigned short* xb  = (unsigned short*)d_ws;               // T x C     bf16
    unsigned short* wqT = xb  + (size_t)T * C;                 // C x C     bf16 [N][K]
    unsigned short* wkT = wqT + (size_t)C * C;                 // NKV x C   bf16
    unsigned short* wvT = wkT + (size_t)NKV * C;               // NKV x C   bf16
    unsigned short* woT = wvT + (size_t)NKV * C;               // C x C     bf16
    float* Qf = (float*)(woT + (size_t)C * C);                 // T x C     f32
    float* Kf = Qf + (size_t)T * C;                            // T x NKV   f32
    float* Vf = Kf + (size_t)T * NKV;                          // T x NKV   f32
    unsigned short* Qn = (unsigned short*)(Vf + (size_t)T * NKV); // T x C  bf16
    unsigned short* Kn = Qn + (size_t)T * C;                   // T x NKV   bf16
    unsigned short* Vt = Kn + (size_t)T * NKV;                 // NKV x T   bf16
    unsigned short* Yb = Vt + (size_t)NKV * T;                 // T x C     bf16

    // 0) one-time bf16 conversion / weight transposition
    convert_f32_to_bf16<<<4096, 256, 0, stream>>>(x, xb, T * C);
    transpose_f32_to_bf16<<<dim3(C / 64, C / 64),   256, 0, stream>>>(wq, wqT, C, C);
    transpose_f32_to_bf16<<<dim3(NKV / 64, C / 64), 256, 0, stream>>>(wk, wkT, C, NKV);
    transpose_f32_to_bf16<<<dim3(NKV / 64, C / 64), 256, 0, stream>>>(wv, wvT, C, NKV);
    transpose_f32_to_bf16<<<dim3(C / 64, C / 64),   256, 0, stream>>>(wo, woT, C, C);

    // 1) projections (async double-buffered bf16 WMMA GEMM)
    wmma_gemm_bf16<<<dim3(C / GTN, T / GTM),   128, 0, stream>>>(xb, wqT, Qf, T, C, C);
    wmma_gemm_bf16<<<dim3(NKV / GTN, T / GTM), 128, 0, stream>>>(xb, wkT, Kf, T, NKV, C);
    wmma_gemm_bf16<<<dim3(NKV / GTN, T / GTM), 128, 0, stream>>>(xb, wvT, Vf, T, NKV, C);

    // 2) RoPE + RMSNorm -> bf16; V transposed to [d][token] for attention
    rope_rms_kernel<<<(T * 16) / 4, 128, 0, stream>>>(Qf, cs, sn, Qn, 16);
    rope_rms_kernel<<<(T * 4) / 4,  128, 0, stream>>>(Kf, cs, sn, Kn, 4);
    transpose_f32_to_bf16<<<dim3(NKV / 64, T / 64), 256, 0, stream>>>(Vf, Vt, T, NKV);

    // 3) causal flash attention (GQA), bf16 out
    flash_attn_kernel<<<dim3(T / QB, 16), 128, 0, stream>>>(Qn, Kn, Vt, Yb);

    // 4) output projection (fp32 out)
    wmma_gemm_bf16<<<dim3(C / GTN, T / GTM), 128, 0, stream>>>(Yb, woT, out, T, C, C);
}
